// Attention_76003741270238
// MI455X (gfx1250) — compile-verified, matrix operand order checked by reading, not run on hardware
//
#include <hip/hip_runtime.h>
#include <hip/hip_bf16.h>

typedef __attribute__((ext_vector_type(16))) __bf16 v16bf;
typedef __attribute__((ext_vector_type(8)))  __bf16 v8bf;
typedef __attribute__((ext_vector_type(8)))  float  v8f;
typedef __attribute__((ext_vector_type(4)))  float  v4f;

#define NHEAD 6
#define DHEAD 64
#define NSEQ  1024
#define BATCH 16
#define DIMM  384
#define INNER 384

// K-index inside a 16-bit 16x32 A (or 32x16 B) fragment for element e (0..15):
// lanes 0-15 (half=0): K = {0..7, 16..23}; lanes 16-31 (half=1): K = {8..15, 24..31}.
// Each half is two contiguous 8-element runs -> fragments load as 2x b128.
__device__ __forceinline__ int kmap(int e, int half) {
    return e + (e < 8 ? half * 8 : 8 + half * 8);
}

__device__ __forceinline__ v8f wmma_bf16(v16bf a, v16bf b, v8f c) {
    return __builtin_amdgcn_wmma_f32_16x16x32_bf16(false, a, false, b, (short)0, c, false, false);
}

__device__ __forceinline__ v8f vzero8() {
    v8f z = {0.f, 0.f, 0.f, 0.f, 0.f, 0.f, 0.f, 0.f};
    return z;
}

// Fragment loader from a bf16 row (contiguous K): two aligned b128 loads.
__device__ __forceinline__ v16bf load_frag_bf16(const __bf16* p, int half) {
    v8bf lo = *(const v8bf*)(p + half * 8);
    v8bf hi = *(const v8bf*)(p + 16 + half * 8);
    return __builtin_shufflevector(lo, hi, 0, 1, 2, 3, 4, 5, 6, 7,
                                   8, 9, 10, 11, 12, 13, 14, 15);
}

// Fragment loader from an fp32 row: four aligned b128 loads + pack-convert.
__device__ __forceinline__ v16bf load_frag_f32(const float* p, int half) {
    v4f a0 = *(const v4f*)(p + half * 8);
    v4f a1 = *(const v4f*)(p + half * 8 + 4);
    v4f b0 = *(const v4f*)(p + 16 + half * 8);
    v4f b1 = *(const v4f*)(p + 16 + half * 8 + 4);
    v16bf r;
#pragma unroll
    for (int e = 0; e < 4; ++e) {
        r[e]      = (__bf16)a0[e];
        r[e + 4]  = (__bf16)a1[e];
        r[e + 8]  = (__bf16)b0[e];
        r[e + 12] = (__bf16)b1[e];
    }
    return r;
}

// ---------------------------------------------------------------------------
// Packed B-operand layout ("B-pack"): for a (Kdim x C) operand consumed as
// WMMA B in 32x16 tiles, tile (kt, ct) is stored fragment-linear:
//   packed[tilebase + lane*16 + e] = W[kt*32 + kmap(e, lane>>4)][ct*16 + (lane&15)]
// so each lane's whole fragment is one contiguous 32B (two b128 loads).
// ---------------------------------------------------------------------------

// Prep kernel: pack fused [Wq | Wkv] (fp32, Kdim=384, C=1152) -> bf16 B-pack.
__global__ __launch_bounds__(32)
void pack_qkv_w(const float* __restrict__ Wq, const float* __restrict__ Wkv,
                __bf16* __restrict__ PW) {
    const int lane = threadIdx.x & 31;
    const int half = lane >> 4;
    const int n    = lane & 15;
    const int ct   = blockIdx.x;              // 0..71
    const int kt   = blockIdx.y;              // 0..11
    const int c    = ct * 16 + n;
    v16bf frag;
    if (c < INNER) {                          // uniform per block
#pragma unroll
        for (int e = 0; e < 16; ++e)
            frag[e] = (__bf16)Wq[(size_t)(kt * 32 + kmap(e, half)) * INNER + c];
    } else {
#pragma unroll
        for (int e = 0; e < 16; ++e)
            frag[e] = (__bf16)Wkv[(size_t)(kt * 32 + kmap(e, half)) * (2 * INNER) + (c - INNER)];
    }
    *(v16bf*)(PW + ((size_t)(ct * (DIMM / 32) + kt) * 32 + lane) * 16) = frag;
}

// Prep kernel: pack Wout (fp32, 384x384) -> bf16 B-pack.
__global__ __launch_bounds__(32)
void pack_out_w(const float* __restrict__ Wout, __bf16* __restrict__ PW) {
    const int lane = threadIdx.x & 31;
    const int half = lane >> 4;
    const int n    = lane & 15;
    const int ct   = blockIdx.x;              // 0..23
    const int kt   = blockIdx.y;              // 0..11
    v16bf frag;
#pragma unroll
    for (int e = 0; e < 16; ++e)
        frag[e] = (__bf16)Wout[(size_t)(kt * 32 + kmap(e, half)) * DIMM + ct * 16 + n];
    *(v16bf*)(PW + ((size_t)(ct * (INNER / 32) + kt) * 32 + lane) * 16) = frag;
}

// ---------------------------------------------------------------------------
// Kernel 1: [Q|K|V] = x @ [Wq|Wkv] via packed weights.  Q,K stored (b,h,n,d)
// bf16 (their attention fragments are contiguous-chunk loadable); V scattered
// directly into B-pack order for the attn.V WMMA (pack once, stream 64x).
// ---------------------------------------------------------------------------
__global__ __launch_bounds__(32)
void qkv_kernel(const float* __restrict__ x, const __bf16* __restrict__ PW,
                __bf16* __restrict__ Q, __bf16* __restrict__ K,
                __bf16* __restrict__ PV) {
    const int lane = threadIdx.x & 31;
    const int half = lane >> 4;
    const int m    = lane & 15;
    const int n    = lane & 15;
    const int col0 = blockIdx.x * 16;   // 0..1151
    const int row0 = blockIdx.y * 16;   // 0..16383 (flat B*N)

    v8f acc = vzero8();
    const float*  xrow = x + (size_t)(row0 + m) * DIMM;
    const __bf16* wp   = PW + ((size_t)blockIdx.x * (DIMM / 32) * 32 + lane) * 16;
    for (int k0 = 0; k0 < DIMM; k0 += 32) {
        v16bf a = load_frag_f32(xrow + k0, half);
        v16bf b = *(const v16bf*)wp;
        wp += 512;                      // next k-tile of this column tile
        acc = wmma_bf16(a, b, acc);
    }

    // C-frag element map: VGPR r -> M = r + half*8, N = lane&15.
    const int c  = col0 + n;
    const int cc = c % INNER;
    const int h  = cc / DHEAD;
    const int d  = cc % DHEAD;
    if (col0 < INNER) {                 // uniform: Q tile
#pragma unroll
        for (int r = 0; r < 8; ++r) {
            const int R = row0 + r + half * 8;
            Q[(((size_t)(R >> 10) * NHEAD + h) * NSEQ + (R & 1023)) * DHEAD + d] = (__bf16)acc[r];
        }
    } else if (col0 < 2 * INNER) {      // uniform: K tile
#pragma unroll
        for (int r = 0; r < 8; ++r) {
            const int R = row0 + r + half * 8;
            K[(((size_t)(R >> 10) * NHEAD + h) * NSEQ + (R & 1023)) * DHEAD + d] = (__bf16)acc[r];
        }
    } else {                            // uniform: V tile -> B-pack scatter
#pragma unroll
        for (int r = 0; r < 8; ++r) {
            const int R   = row0 + r + half * 8;
            const int bb  = R >> 10;
            const int j   = R & 1023;
            const int jt2 = j >> 5;
            const int jl  = j & 31;
            const int hp  = (jl >> 3) & 1;                    // fragment half
            const int ep  = (jl & 7) + (jl >= 16 ? 8 : 0);    // element in half
            const int lp  = hp * 16 + (d & 15);               // lane in tile
            const size_t base = (size_t)(bb * NHEAD + h) * (NSEQ * DHEAD);
            PV[base + ((size_t)(jt2 * 4 + (d >> 4)) * 32 + lp) * 16 + ep] = (__bf16)acc[r];
        }
    }
}

// ---------------------------------------------------------------------------
// Score helper: raw dots (all 6 heads) for one 16(i) x 16(j) tile,
// per-head scale, LSA diagonal forced to -1e-9 (exactly as reference).
// ---------------------------------------------------------------------------
__device__ __forceinline__ void compute_scores(
        const __bf16* __restrict__ Q, const __bf16* __restrict__ K,
        int b, int i0, int jj, const float* sc,
        int half, int m, int n, float (*raws)[8]) {
#pragma unroll
    for (int h = 0; h < NHEAD; ++h) {
        const __bf16* qh = Q + (((size_t)b * NHEAD + h) * NSEQ + i0) * DHEAD + (size_t)m * DHEAD;
        const __bf16* kh = K + (((size_t)b * NHEAD + h) * NSEQ + jj) * DHEAD + (size_t)n * DHEAD;
        v8f acc = vzero8();
#pragma unroll
        for (int kb = 0; kb < DHEAD; kb += 32) {
            v16bf a  = load_frag_bf16(qh + kb, half);   // A: 16(i) x 32(d)
            v16bf bb = load_frag_bf16(kh + kb, half);   // B: 32(d) x 16(j) from K^T
            acc = wmma_bf16(a, bb, acc);
        }
        const float s = sc[h];
#pragma unroll
        for (int r = 0; r < 8; ++r) {
            const int R = i0 + r + half * 8;
            const int C = jj + n;
            raws[h][r] = (R == C) ? -1e-9f : acc[r] * s;
        }
    }
}

// ---------------------------------------------------------------------------
// Kernel 2: fused talking-heads attention, two-pass (recompute) softmax.
// One wave owns (batch b, 16 query rows, ALL 6 heads) since pre/post head
// mixing couples heads at each (i,j).
// ---------------------------------------------------------------------------
__global__ __launch_bounds__(32)
void attn_kernel(const __bf16* __restrict__ Q, const __bf16* __restrict__ K,
                 const __bf16* __restrict__ PV,
                 const float* __restrict__ scale,
                 const float* __restrict__ mix_pre,
                 const float* __restrict__ mix_post,
                 __bf16* __restrict__ AO) {
    __shared__ __bf16 sA[NHEAD * 16 * 32];   // mixed probs staged for C->A relayout

    const int lane = threadIdx.x & 31;
    const int half = lane >> 4;
    const int m    = lane & 15;
    const int n    = lane & 15;
    const int b    = blockIdx.x >> 6;
    const int i0   = (blockIdx.x & 63) * 16;

    float sc[NHEAD], pre[NHEAD][NHEAD], post[NHEAD][NHEAD];
#pragma unroll
    for (int h = 0; h < NHEAD; ++h) sc[h] = scale[h];
#pragma unroll
    for (int h = 0; h < NHEAD; ++h)
#pragma unroll
        for (int g = 0; g < NHEAD; ++g) {
            pre[h][g]  = mix_pre[h * NHEAD + g];    // 'bhij,hg->bgij'
            post[h][g] = mix_post[h * NHEAD + g];
        }

    // ---- Pass 1: per-(head,row) online max & exp-sum over all 1024 keys.
    float mr[NHEAD][8], lr[NHEAD][8];
#pragma unroll
    for (int g = 0; g < NHEAD; ++g)
#pragma unroll
        for (int r = 0; r < 8; ++r) { mr[g][r] = -3.0e38f; lr[g][r] = 0.f; }

    for (int jt = 0; jt < NSEQ / 16; ++jt) {
        __builtin_prefetch(K + (((size_t)b * NHEAD) * NSEQ + (jt + 1) * 16) * DHEAD, 0, 1);
        float raws[NHEAD][8];
        compute_scores(Q, K, b, i0, jt * 16, sc, half, m, n, raws);
#pragma unroll
        for (int g = 0; g < NHEAD; ++g)
#pragma unroll
            for (int r = 0; r < 8; ++r) {
                float mx = 0.f;
#pragma unroll
                for (int h = 0; h < NHEAD; ++h) mx += pre[h][g] * raws[h][r];
                const float mnew = fmaxf(mr[g][r], mx);
                lr[g][r] = lr[g][r] * __expf(mr[g][r] - mnew) + __expf(mx - mnew);
                mr[g][r] = mnew;
            }
    }
    // Cross-lane combine: lanes l, l^mask (mask<16) share the same row M.
#pragma unroll
    for (int g = 0; g < NHEAD; ++g)
#pragma unroll
        for (int r = 0; r < 8; ++r) {
            float mv = mr[g][r], lv = lr[g][r];
#pragma unroll
            for (int mask = 1; mask < 16; mask <<= 1) {
                const float mo = __shfl_xor(mv, mask, 32);
                const float lo = __shfl_xor(lv, mask, 32);
                const float mn = fmaxf(mv, mo);
                lv = lv * __expf(mv - mn) + lo * __expf(mo - mn);
                mv = mn;
            }
            mr[g][r] = mv; lr[g][r] = lv;
        }

    // ---- Pass 2: recompute scores, P = exp(s-m)/Z, post-mix, out += A @ V.
    v8f oacc[NHEAD][4];
#pragma unroll
    for (int g = 0; g < NHEAD; ++g)
#pragma unroll
        for (int dt = 0; dt < 4; ++dt) oacc[g][dt] = vzero8();

    for (int jt2 = 0; jt2 < NSEQ / 32; ++jt2) {
        const int j0 = jt2 * 32;
#pragma unroll
        for (int sub = 0; sub < 2; ++sub) {
            float raws[NHEAD][8];
            compute_scores(Q, K, b, i0, j0 + sub * 16, sc, half, m, n, raws);
            float P[NHEAD][8];
#pragma unroll
            for (int g = 0; g < NHEAD; ++g)
#pragma unroll
                for (int r = 0; r < 8; ++r) {
                    float mx = 0.f;
#pragma unroll
                    for (int h = 0; h < NHEAD; ++h) mx += pre[h][g] * raws[h][r];
                    P[g][r] = __expf(mx - mr[g][r]) / lr[g][r];
                }
#pragma unroll
            for (int gp = 0; gp < NHEAD; ++gp)
#pragma unroll
                for (int r = 0; r < 8; ++r) {
                    float av = 0.f;
#pragma unroll
                    for (int g = 0; g < NHEAD; ++g) av += post[g][gp] * P[g][r];
                    sA[gp * 512 + (r + half * 8) * 32 + sub * 16 + n] = (__bf16)av;
                }
        }
        __syncthreads();
#pragma unroll
        for (int gp = 0; gp < NHEAD; ++gp) {
            v16bf afr = load_frag_bf16(sA + gp * 512 + m * 32, half);  // ds_load b128 x2
            const __bf16* pv = PV + (size_t)(b * NHEAD + gp) * (NSEQ * DHEAD)
                                  + (size_t)(jt2 * 4) * 512 + (size_t)lane * 16;
            __builtin_prefetch(pv + 2048, 0, 1);
#pragma unroll
            for (int dt = 0; dt < 4; ++dt) {
                v16bf bfr = *(const v16bf*)(pv + (size_t)dt * 512);    // contiguous B-pack
                oacc[gp][dt] = wmma_bf16(afr, bfr, oacc[gp][dt]);
            }
        }
        __syncthreads();
    }

    // Store per-head output tile (b, gp, i0+M, d) as bf16.
#pragma unroll
    for (int gp = 0; gp < NHEAD; ++gp)
#pragma unroll
        for (int dt = 0; dt < 4; ++dt)
#pragma unroll
            for (int r = 0; r < 8; ++r) {
                const int nn = i0 + r + half * 8;
                AO[(((size_t)b * NHEAD + gp) * NSEQ + nn) * DHEAD + dt * 16 + n] =
                    (__bf16)oacc[gp][dt][r];
            }
}

// ---------------------------------------------------------------------------
// Kernel 3: out = AO(b,n,(h d)) @ Wout + bout, fp32 result, packed weights.
// ---------------------------------------------------------------------------
__global__ __launch_bounds__(32)
void outproj_kernel(const __bf16* __restrict__ AO, const __bf16* __restrict__ PW,
                    const float* __restrict__ bout, float* __restrict__ out) {
    const int lane = threadIdx.x & 31;
    const int half = lane >> 4;
    const int m    = lane & 15;
    const int n    = lane & 15;
    const int col0 = blockIdx.x * 16;   // 0..383
    const int row0 = blockIdx.y * 16;   // 0..16383

    const int R  = row0 + m;
    const int bb = R >> 10;
    const int nn = R & 1023;

    v8f acc = vzero8();
    const __bf16* wp = PW + ((size_t)blockIdx.x * (INNER / 32) * 32 + lane) * 16;
    for (int k0 = 0; k0 < INNER; k0 += 32) {
        const int h = k0 >> 6;          // 32-wide k-block never straddles a head
        const __bf16* ap = AO + (((size_t)bb * NHEAD + h) * NSEQ + nn) * DHEAD + (k0 & 63);
        v16bf a   = load_frag_bf16(ap, half);
        v16bf bfr = *(const v16bf*)wp;
        wp += 512;
        acc = wmma_bf16(a, bfr, acc);
    }
    const float bias = bout[col0 + n];
#pragma unroll
    for (int r = 0; r < 8; ++r) {
        const int Rr = row0 + r + half * 8;
        out[(size_t)Rr * DIMM + col0 + n] = acc[r] + bias;
    }
}

// ---------------------------------------------------------------------------
extern "C" void kernel_launch(void* const* d_in, const int* in_sizes, int n_in,
                              void* d_out, int out_size, void* d_ws, size_t ws_size,
                              hipStream_t stream) {
    (void)in_sizes; (void)n_in; (void)out_size; (void)ws_size;
    const float* x        = (const float*)d_in[0];
    const float* Wq       = (const float*)d_in[1];
    const float* Wkv      = (const float*)d_in[2];
    const float* scale    = (const float*)d_in[3];
    const float* mix_pre  = (const float*)d_in[4];
    const float* mix_post = (const float*)d_in[5];
    const float* Wout     = (const float*)d_in[6];
    const float* bout     = (const float*)d_in[7];
    float* out            = (float*)d_out;

    // Workspace (bf16): Q, K, PV (packed V), AO, packed weights. ~51.5 MB.
    const size_t elems = (size_t)BATCH * NHEAD * NSEQ * DHEAD;   // 6,291,456
    __bf16* Q     = (__bf16*)d_ws;
    __bf16* K     = Q + elems;
    __bf16* PV    = K + elems;
    __bf16* AO    = PV + elems;
    __bf16* PWqkv = AO + elems;                                  // 384*1152
    __bf16* PWout = PWqkv + (size_t)DIMM * (3 * INNER);          // 384*384

    dim3 blk(32, 1, 1);
    // 0) weight packing (tiny, once per launch; stays hot in L2)
    pack_qkv_w<<<dim3(3 * INNER / 16, DIMM / 32), blk, 0, stream>>>(Wq, Wkv, PWqkv);
    pack_out_w<<<dim3(DIMM / 16, INNER / 32), blk, 0, stream>>>(Wout, PWout);
    // 1) fused QKV projection: 72 col-tiles x 1024 row-tiles
    qkv_kernel<<<dim3(3 * INNER / 16, BATCH * NSEQ / 16), blk, 0, stream>>>(
        x, PWqkv, Q, K, PV);
    // 2) fused talking-heads attention: one wave per (b, 16-row tile)
    attn_kernel<<<dim3(BATCH * NSEQ / 16), blk, 0, stream>>>(
        Q, K, PV, scale, mix_pre, mix_post, AO);
    // 3) output projection + bias
    outproj_kernel<<<dim3(DIMM / 16, BATCH * NSEQ / 16), blk, 0, stream>>>(
        AO, PWout, bout, out);
}
